// MambaHead_3573412790308
// MI455X (gfx1250) — compile-verified, matrix operand order checked by reading, not run on hardware
//
#include <hip/hip_runtime.h>
#include <math.h>

// ---------------------------------------------------------------------------
// Types for CDNA5 WMMA / TDM (gfx1250, wave32)
// ---------------------------------------------------------------------------
typedef __attribute__((ext_vector_type(16))) _Float16 v16h;
typedef __attribute__((ext_vector_type(8)))  float    v8f;
typedef __attribute__((ext_vector_type(4)))  unsigned int v4u;
typedef __attribute__((ext_vector_type(8)))  int      v8i;
typedef __attribute__((ext_vector_type(4)))  int      v4i;

#ifndef __has_builtin
#define __has_builtin(x) 0
#endif

#if __has_builtin(__builtin_amdgcn_tensor_load_to_lds)
#define USE_TDM 1
#else
#define USE_TDM 0
#endif

union FragU { v16h h; uint4 u[2]; };

__device__ __forceinline__ v16h load_frag16(const _Float16* rowp, int klo, int khi) {
    FragU f;
    f.u[0] = *reinterpret_cast<const uint4*>(rowp + klo);
    f.u[1] = *reinterpret_cast<const uint4*>(rowp + khi);
    return f.h;
}

__device__ __forceinline__ v8f wmma_f16(v16h a, v16h b, v8f c) {
    // emits v_wmma_f32_16x16x32_f16
    return __builtin_amdgcn_wmma_f32_16x16x32_f16(
        /*neg_a=*/false, a, /*neg_b=*/false, b,
        /*c_mod=*/(short)0, c, /*reuse_a=*/false, /*reuse_b=*/false);
}

__device__ __forceinline__ float silu_f(float x) {
    // v_rcp_f32 instead of IEEE divide chain (feeds f16 WMMA anyway)
    return x * __builtin_amdgcn_rcpf(1.0f + __expf(-x));
}

// Issue a TDM load of a 2D tile (rows x kElems f16, row stride kElems) into LDS.
// Descriptor packed per CDNA5 ISA section 8.3/8.4 (D# groups 0/1; groups 2/3 zero
// for a 2D tile).
__device__ __forceinline__ void tdm_load_tile_f16(unsigned ldsOff,
                                                  unsigned long long gaddr,
                                                  int kElems, int rows) {
#if USE_TDM
    v4u g0 = {
        1u,                                            // count=1 (valid descriptor)
        ldsOff,                                        // lds_addr
        (unsigned)(gaddr & 0xFFFFFFFFull),             // global_addr[31:0]
        (unsigned)((gaddr >> 32) & 0x01FFFFFFull) | (2u << 30)  // addr[56:32] | type=2
    };
    unsigned uk = (unsigned)kElems, ur = (unsigned)rows;
    v8i g1 = {
        (int)(1u << 16),                 // workgroup_mask=0, data_size=1 (2 bytes)
        (int)((uk & 0xFFFFu) << 16),     // tensor_dim0[15:0]
        (int)(((uk >> 16) & 0xFFFFu) | (ur << 16)),  // tensor_dim0[31:16] | tensor_dim1[15:0]
        (int)(uk << 16),                 // tensor_dim1[31:16]=0 | tile_dim0
        (int)ur,                         // tile_dim1 | tile_dim2=0
        (int)uk,                         // tensor_dim0_stride[31:0]
        0,                               // stride[47:32] | tensor_dim1_stride[15:0]
        0                                // tensor_dim1_stride[47:16]
    };
    v4i gz = {0, 0, 0, 0};
#if defined(__clang_major__) && (__clang_major__ >= 23)
    v8i gz8 = {0, 0, 0, 0, 0, 0, 0, 0};
    __builtin_amdgcn_tensor_load_to_lds(g0, g1, gz, gz, gz8, 0);
#else
    __builtin_amdgcn_tensor_load_to_lds(g0, g1, gz, gz, 0);
#endif
#if __has_builtin(__builtin_amdgcn_s_wait_tensorcnt)
    __builtin_amdgcn_s_wait_tensorcnt(0);
#endif
#else
    (void)ldsOff; (void)gaddr; (void)kElems; (void)rows;
#endif
}

// Cubic B-spline basis, grid_size=5, order=3, range [-1,1] (efficient-KAN layout).
// Knots t_i = -2.2 + 0.4*i, i=0..11 ; produces 8 coefficients.
__device__ __forceinline__ void bspline8(float x, float* o) {
    const float hh = 0.4f;
    float b[11];
#pragma unroll
    for (int i = 0; i < 11; ++i) {
        float t = -2.2f + hh * i;
        b[i] = (x >= t && x < t + hh) ? 1.0f : 0.0f;
    }
#pragma unroll
    for (int kk = 1; kk <= 3; ++kk) {
        float inv = 1.0f / (hh * kk);
#pragma unroll
        for (int i = 0; i < 11 - kk; ++i) {
            float ti  = -2.2f + hh * i;
            float tk1 = ti + hh * (kk + 1);
            b[i] = (x - ti) * inv * b[i] + (tk1 - x) * inv * b[i + 1];
        }
    }
#pragma unroll
    for (int i = 0; i < 8; ++i) o[i] = b[i];
}

// ---------------------------------------------------------------------------
// Single-wave WMMA GEMM:  C[M,N] (f32) = A[M,K] (f16 row-major) * W[N,K]^T (f16)
// Used for the small-N x_proj GEMM.
// ---------------------------------------------------------------------------
template <int NT, bool ACC>
__global__ __launch_bounds__(32) void gemm_wmma_kernel(
    const _Float16* __restrict__ A, const _Float16* __restrict__ W,
    float* __restrict__ C, int M, int N, int K) {
    const int lane = threadIdx.x;
    const int lm = lane & 15;
    const int hi = lane >> 4;
    const int mt = M >> 4;
    const int tm = blockIdx.x % mt;
    const int ng = blockIdx.x / mt;
    const int row0 = tm << 4;
    const int col0 = ng * (16 * NT);

    v8f acc[NT];
#pragma unroll
    for (int t = 0; t < NT; ++t) {
#pragma unroll
        for (int r = 0; r < 8; ++r) {
            if (ACC)
                acc[t][r] = C[(size_t)(row0 + r + hi * 8) * N + (col0 + t * 16 + lm)];
            else
                acc[t][r] = 0.0f;
        }
    }

    const _Float16* arow = A + (size_t)(row0 + lm) * K;
    for (int k0 = 0; k0 < K; k0 += 32) {
        const int klo = k0 + hi * 8;
        const int khi = k0 + 16 + hi * 8;
        __builtin_prefetch(arow + k0 + 128, 0, 0);   // global_prefetch_b8
        v16h a = load_frag16(arow, klo, khi);
#pragma unroll
        for (int t = 0; t < NT; ++t) {
            const _Float16* wrow = W + (size_t)(col0 + t * 16 + lm) * K;
            v16h b = load_frag16(wrow, klo, khi);
            acc[t] = wmma_f16(a, b, acc[t]);
        }
    }

#pragma unroll
    for (int t = 0; t < NT; ++t)
#pragma unroll
        for (int r = 0; r < 8; ++r)
            C[(size_t)(row0 + r + hi * 8) * N + (col0 + t * 16 + lm)] = acc[t][r];
}

// ---------------------------------------------------------------------------
// Multi-wave WMMA GEMM with the A-panel staged in LDS by the Tensor Data Mover.
// Block = NW waves; the 16 x K A-panel is loaded ONCE (TDM async-DMA, wave 0),
// then all NW waves consume it from LDS (ds_load_b128 fragments) against their
// own 16*NT-column slice of W. K <= 512.
// ---------------------------------------------------------------------------
template <int NT, int NW, bool ACC>
__global__ __launch_bounds__(NW * 32) void gemm_wmma_lds_kernel(
    const _Float16* __restrict__ A, const _Float16* __restrict__ W,
    float* __restrict__ C, int M, int N, int K) {
    __shared__ _Float16 tileA[16 * 512];
    const int tid = threadIdx.x;
    const int lane = tid & 31;
    const int wv = tid >> 5;
    const int lm = lane & 15;
    const int hi = (lane >> 4) & 1;
    const int mt = M >> 4;
    const int tm = blockIdx.x % mt;
    const int ng = blockIdx.x / mt;
    const int row0 = tm << 4;
    const int col0 = ng * (16 * NT * NW) + wv * (16 * NT);

#if USE_TDM
    if (wv == 0) {
        unsigned ldsOff = (unsigned)(unsigned long long)(uintptr_t)&tileA[0];
        unsigned long long ga =
            (unsigned long long)(uintptr_t)A + (unsigned long long)row0 * K * 2ull;
        tdm_load_tile_f16(ldsOff, ga, K, 16);   // tensor_load_to_lds + s_wait_tensorcnt
    }
#else
    for (int idx = tid; idx < (16 * K) / 8; idx += NW * 32) {
        const int r = idx / (K / 8);
        const int kk = (idx % (K / 8)) * 8;
        *reinterpret_cast<uint4*>(&tileA[r * K + kk]) =
            *reinterpret_cast<const uint4*>(A + (size_t)(row0 + r) * K + kk);
    }
#endif
    __syncthreads();

    v8f acc[NT];
#pragma unroll
    for (int t = 0; t < NT; ++t) {
#pragma unroll
        for (int r = 0; r < 8; ++r) {
            if (ACC)
                acc[t][r] = C[(size_t)(row0 + r + hi * 8) * N + (col0 + t * 16 + lm)];
            else
                acc[t][r] = 0.0f;
        }
    }

    const _Float16* arow = &tileA[lm * K];
    for (int k0 = 0; k0 < K; k0 += 32) {
        const int klo = k0 + hi * 8;
        const int khi = k0 + 16 + hi * 8;
        v16h a = load_frag16(arow, klo, khi);   // ds_load_b128 pair from LDS
#pragma unroll
        for (int t = 0; t < NT; ++t) {
            const _Float16* wrow = W + (size_t)(col0 + t * 16 + lm) * K;
            __builtin_prefetch(wrow + k0 + 128, 0, 0);
            v16h b = load_frag16(wrow, klo, khi);
            acc[t] = wmma_f16(a, b, acc[t]);
        }
    }

#pragma unroll
    for (int t = 0; t < NT; ++t)
#pragma unroll
        for (int r = 0; r < 8; ++r)
            C[(size_t)(row0 + r + hi * 8) * N + (col0 + t * 16 + lm)] = acc[t][r];
}

// ---------------------------------------------------------------------------
// Fused KAN-linear #1: h[8192,256] = silu(x) @ bw^T + bspline(x) @ swc^T
// A-fragments built in registers: base part K=768, spline part K=768*8=6144
// (each 8-half run of the fragment == 8 spline coeffs of one input feature).
// ---------------------------------------------------------------------------
__global__ __launch_bounds__(32) void kan1_wmma_kernel(
    const float* __restrict__ x, const _Float16* __restrict__ bw,
    const _Float16* __restrict__ sw, float* __restrict__ hout) {
    const int lane = threadIdx.x;
    const int lm = lane & 15;
    const int hi = lane >> 4;
    const int tm = blockIdx.x & 511;   // 8192/16 m-tiles
    const int ng = blockIdx.x >> 9;    // 4 n-groups of 64 cols
    const int row0 = tm << 4;
    const int col0 = ng << 6;
    const float* xrow = x + (size_t)(row0 + lm) * 768;

    v8f acc[4];
#pragma unroll
    for (int t = 0; t < 4; ++t)
#pragma unroll
        for (int r = 0; r < 8; ++r) acc[t][r] = 0.0f;

    // ---- base branch: A[m,k] = silu(x[m,k]), K = 768
    for (int k0 = 0; k0 < 768; k0 += 32) {
        const int klo = k0 + hi * 8;
        const int khi = k0 + 16 + hi * 8;
        v16h a;
#pragma unroll
        for (int j = 0; j < 8; ++j) {
            a[j]     = (_Float16)silu_f(xrow[klo + j]);
            a[8 + j] = (_Float16)silu_f(xrow[khi + j]);
        }
#pragma unroll
        for (int t = 0; t < 4; ++t) {
            const _Float16* wrow = bw + (size_t)(col0 + t * 16 + lm) * 768;
            acc[t] = wmma_f16(a, load_frag16(wrow, klo, khi), acc[t]);
        }
    }

    // ---- spline branch: A[m, d*8+c] = B_c(x[m,d]), K = 6144
    for (int ks = 0; ks < 6144; ks += 32) {
        const int d0 = ks >> 3;        // 4 consecutive features per k-chunk
        float xa = xrow[d0 + hi];      // features d0 / d0+1
        float xb = xrow[d0 + 2 + hi];  // features d0+2 / d0+3
        float ba[8], bb[8];
        bspline8(xa, ba);
        bspline8(xb, bb);
        v16h a;
#pragma unroll
        for (int j = 0; j < 8; ++j) {
            a[j]     = (_Float16)ba[j];
            a[8 + j] = (_Float16)bb[j];
        }
        const int klo = ks + hi * 8;
        const int khi = ks + 16 + hi * 8;
#pragma unroll
        for (int t = 0; t < 4; ++t) {
            const _Float16* wrow = sw + (size_t)(col0 + t * 16 + lm) * 6144;
            acc[t] = wmma_f16(a, load_frag16(wrow, klo, khi), acc[t]);
        }
    }

#pragma unroll
    for (int t = 0; t < 4; ++t)
#pragma unroll
        for (int r = 0; r < 8; ++r)
            hout[(size_t)(row0 + r + hi * 8) * 256 + col0 + t * 16 + lm] = acc[t][r];
}

// ---------------------------------------------------------------------------
// Norms (one 256-thread block per row, LDS tree reduction)
// ---------------------------------------------------------------------------
__global__ void layernorm_kernel(float* __restrict__ h, const float* __restrict__ g,
                                 const float* __restrict__ b) {
    __shared__ float s1[256];
    __shared__ float s2[256];
    const int row = blockIdx.x, c = threadIdx.x;
    float v = h[(size_t)row * 256 + c];
    s1[c] = v; s2[c] = v * v;
    __syncthreads();
    for (int st = 128; st > 0; st >>= 1) {
        if (c < st) { s1[c] += s1[c + st]; s2[c] += s2[c + st]; }
        __syncthreads();
    }
    float mean = s1[0] * (1.0f / 256.0f);
    float var  = s2[0] * (1.0f / 256.0f) - mean * mean;
    h[(size_t)row * 256 + c] = (v - mean) * rsqrtf(var + 1e-5f) * g[c] + b[c];
}

__global__ void rmsnorm16_kernel(const float* __restrict__ h, const float* __restrict__ w,
                                 _Float16* __restrict__ out) {
    __shared__ float s1[256];
    const int row = blockIdx.x, c = threadIdx.x;
    float v = h[(size_t)row * 256 + c];
    s1[c] = v * v;
    __syncthreads();
    for (int st = 128; st > 0; st >>= 1) {
        if (c < st) s1[c] += s1[c + st];
        __syncthreads();
    }
    float ms = s1[0] * (1.0f / 256.0f);
    out[(size_t)row * 256 + c] = (_Float16)(w[c] * v * rsqrtf(ms + 1e-5f));
}

__global__ void rmsnormf_kernel(const float* __restrict__ h, const float* __restrict__ w,
                                float* __restrict__ out) {
    __shared__ float s1[256];
    const int row = blockIdx.x, c = threadIdx.x;
    float v = h[(size_t)row * 256 + c];
    s1[c] = v * v;
    __syncthreads();
    for (int st = 128; st > 0; st >>= 1) {
        if (c < st) s1[c] += s1[c + st];
        __syncthreads();
    }
    float ms = s1[0] * (1.0f / 256.0f);
    out[(size_t)row * 256 + c] = w[c] * v * rsqrtf(ms + 1e-5f);
}

// ---------------------------------------------------------------------------
// Mamba elementwise stages
// ---------------------------------------------------------------------------
__global__ void conv_silu_kernel(const float* __restrict__ proj, const float* __restrict__ cw,
                                 const float* __restrict__ cb, float* __restrict__ uc,
                                 _Float16* __restrict__ uc16, int total) {
    int idx = blockIdx.x * blockDim.x + threadIdx.x;
    if (idx >= total) return;
    const int i = idx & 511;
    const int m = idx >> 9;
    const int l = m & 511;            // position within sequence
    float s = cb[i];
#pragma unroll
    for (int kk = 0; kk < 4; ++kk) {
        int lp = l - 3 + kk;
        if (lp >= 0) s += proj[(size_t)(m - 3 + kk) * 1024 + i] * cw[i * 4 + kk];
    }
    float v = silu_f(s);
    uc[idx] = v;
    uc16[idx] = (_Float16)v;
}

__global__ void dt_kernel(const float* __restrict__ ssm, const float* __restrict__ dtw,
                          const float* __restrict__ dtb, float* __restrict__ dt, int total) {
    int idx = blockIdx.x * blockDim.x + threadIdx.x;
    if (idx >= total) return;
    const int i = idx & 511;
    const int m = idx >> 9;
    float s = dtb[i];
    const float* r = ssm + (size_t)m * 48;
#pragma unroll
    for (int j = 0; j < 16; ++j) s += r[j] * dtw[i * 16 + j];
    dt[idx] = (s > 20.0f) ? s : log1pf(__expf(s));   // softplus
}

// Selective scan: one thread per (batch, channel), 16-entry state in registers.
__global__ void scan_kernel(const float* __restrict__ dt, const float* __restrict__ uc,
                            const float* __restrict__ ssm, const float* __restrict__ alog,
                            float* __restrict__ ys) {
    int t = blockIdx.x * blockDim.x + threadIdx.x;   // 8192 threads
    const int b = t >> 9, i = t & 511;
    float A[16];
#pragma unroll
    for (int n = 0; n < 16; ++n) A[n] = -__expf(alog[i * 16 + n]);
    float st[16];
#pragma unroll
    for (int n = 0; n < 16; ++n) st[n] = 0.0f;
    for (int l = 0; l < 512; ++l) {
        const int m = (b << 9) | l;
        float dtv = dt[(size_t)m * 512 + i];
        float uv  = uc[(size_t)m * 512 + i];
        float du = dtv * uv;
        const float* sr = ssm + (size_t)m * 48;
        float y = 0.0f;
#pragma unroll
        for (int n = 0; n < 16; ++n) {
            st[n] = st[n] * __expf(dtv * A[n]) + du * sr[16 + n];
            y += st[n] * sr[32 + n];
        }
        ys[(size_t)m * 512 + i] = y;
    }
}

__global__ void gate_kernel(const float* __restrict__ ys, const float* __restrict__ uc,
                            const float* __restrict__ proj, const float* __restrict__ Dp,
                            _Float16* __restrict__ y16, int total) {
    int idx = blockIdx.x * blockDim.x + threadIdx.x;
    if (idx >= total) return;
    const int i = idx & 511;
    const int m = idx >> 9;
    float g = proj[(size_t)m * 1024 + 512 + i];
    float y = (ys[idx] + uc[idx] * Dp[i]) * silu_f(g);
    y16[idx] = (_Float16)y;
}

// ---------------------------------------------------------------------------
// Pooling + tiny KAN head
// ---------------------------------------------------------------------------
__global__ void pool_kernel(const float* __restrict__ hf, float* __restrict__ emb) {
    int t = blockIdx.x * blockDim.x + threadIdx.x;
    if (t >= 4096) return;
    const int b = t >> 8, c = t & 255;
    float s = 0.0f, mx = -INFINITY;
    for (int l = 0; l < 512; ++l) {
        float v = hf[(size_t)((b << 9) | l) * 256 + c];
        s += v;
        mx = fmaxf(mx, v);
    }
    emb[b * 512 + c]       = s * (1.0f / 512.0f);
    emb[b * 512 + 256 + c] = mx;
}

__global__ void kan2_kernel(const float* __restrict__ emb, const float* __restrict__ bw,
                            const float* __restrict__ sw, const float* __restrict__ ss,
                            float* __restrict__ logits) {
    int t = threadIdx.x;
    if (t >= 32) return;
    const int b = t >> 1, o = t & 1;
    const float* e = emb + b * 512;
    float s = 0.0f;
    for (int j = 0; j < 512; ++j) {
        float v = e[j];
        s += silu_f(v) * bw[o * 512 + j];
        float bs[8];
        bspline8(v, bs);
        float sc = ss[o * 512 + j];
        const float* w = sw + (size_t)(o * 512 + j) * 8;
#pragma unroll
        for (int c = 0; c < 8; ++c) s += bs[c] * w[c] * sc;
    }
    logits[b * 2 + o] = s;
}

// ---------------------------------------------------------------------------
// Weight conversion kernels
// ---------------------------------------------------------------------------
__global__ void cvt16_kernel(const float* __restrict__ in, _Float16* __restrict__ out, int n) {
    int idx = blockIdx.x * blockDim.x + threadIdx.x;
    if (idx < n) out[idx] = (_Float16)in[idx];
}

__global__ void spline_combine_kernel(const float* __restrict__ sw, const float* __restrict__ ss,
                                      _Float16* __restrict__ out, int n) {
    int idx = blockIdx.x * blockDim.x + threadIdx.x;
    if (idx < n) out[idx] = (_Float16)(sw[idx] * ss[idx >> 3]);
}

// ---------------------------------------------------------------------------
// Host-side orchestration
// ---------------------------------------------------------------------------
extern "C" void kernel_launch(void* const* d_in, const int* in_sizes, int n_in,
                              void* d_out, int out_size, void* d_ws, size_t ws_size,
                              hipStream_t stream) {
    (void)in_sizes; (void)n_in; (void)out_size; (void)ws_size;

    const float* x         = (const float*)d_in[0];
    const float* base_w1   = (const float*)d_in[1];
    const float* spline_w1 = (const float*)d_in[2];
    const float* spline_s1 = (const float*)d_in[3];
    const float* ln_g      = (const float*)d_in[4];
    const float* ln_b      = (const float*)d_in[5];
    const float* norm_w    = (const float*)d_in[6];
    const float* in_proj_w = (const float*)d_in[7];
    const float* conv_w    = (const float*)d_in[8];
    const float* conv_b    = (const float*)d_in[9];
    const float* x_proj_w  = (const float*)d_in[10];
    const float* dt_w      = (const float*)d_in[11];
    const float* dt_b      = (const float*)d_in[12];
    const float* A_log     = (const float*)d_in[13];
    const float* Dp        = (const float*)d_in[14];
    const float* out_proj_w= (const float*)d_in[15];
    const float* normf_w   = (const float*)d_in[16];
    const float* base_w2   = (const float*)d_in[17];
    const float* spline_w2 = (const float*)d_in[18];
    const float* spline_s2 = (const float*)d_in[19];

    const int M = 16 * 512;   // B*L = 8192 rows

    // workspace carve (all offsets 256B aligned)
    char* base = (char*)d_ws;
    size_t off = 0;
    auto carve = [&](size_t bytes) -> void* {
        void* r = base + off;
        off = (off + bytes + 255) & ~(size_t)255;
        return r;
    };
    _Float16* bw1_16    = (_Float16*)carve((size_t)256 * 768 * 2);
    _Float16* swc16     = (_Float16*)carve((size_t)256 * 768 * 8 * 2);
    _Float16* inproj16  = (_Float16*)carve((size_t)2 * 1024 * 256 * 2);
    _Float16* xproj16   = (_Float16*)carve((size_t)2 * 48 * 512 * 2);
    _Float16* outproj16 = (_Float16*)carve((size_t)2 * 256 * 512 * 2);
    float*    h         = (float*)carve((size_t)M * 256 * 4);
    _Float16* hn16      = (_Float16*)carve((size_t)M * 256 * 2);
    float*    proj      = (float*)carve((size_t)M * 1024 * 4);
    float*    uc        = (float*)carve((size_t)M * 512 * 4);
    _Float16* uc16      = (_Float16*)carve((size_t)M * 512 * 2);
    float*    ssm       = (float*)carve((size_t)M * 48 * 4);
    float*    dtb_      = (float*)carve((size_t)M * 512 * 4);
    float*    ysb       = (float*)carve((size_t)M * 512 * 4);
    _Float16* y16       = (_Float16*)carve((size_t)M * 512 * 2);
    float*    hf        = (float*)carve((size_t)M * 256 * 4);

    const int TB = 256;
    auto g1 = [&](int n) { return dim3((n + TB - 1) / TB); };

    // weight conversions (per-call, deterministic)
    cvt16_kernel<<<g1(256 * 768), TB, 0, stream>>>(base_w1, bw1_16, 256 * 768);
    spline_combine_kernel<<<g1(256 * 768 * 8), TB, 0, stream>>>(spline_w1, spline_s1, swc16, 256 * 768 * 8);
    cvt16_kernel<<<g1(2 * 1024 * 256), TB, 0, stream>>>(in_proj_w, inproj16, 2 * 1024 * 256);
    cvt16_kernel<<<g1(2 * 48 * 512), TB, 0, stream>>>(x_proj_w, xproj16, 2 * 48 * 512);
    cvt16_kernel<<<g1(2 * 256 * 512), TB, 0, stream>>>(out_proj_w, outproj16, 2 * 256 * 512);

    // KAN-1 (fused base+spline WMMA GEMM) -> h, then layernorm
    kan1_wmma_kernel<<<dim3(512 * 4), dim3(32), 0, stream>>>(x, bw1_16, swc16, h);
    layernorm_kernel<<<dim3(M), dim3(256), 0, stream>>>(h, ln_g, ln_b);

    // 2 residual Mamba blocks
    for (int i = 0; i < 2; ++i) {
        rmsnorm16_kernel<<<dim3(M), dim3(256), 0, stream>>>(h, norm_w + i * 256, hn16);
        // in_proj: 4 waves share one TDM-staged A panel; each block -> 16x256 tile
        gemm_wmma_lds_kernel<4, 4, false><<<dim3(512 * 4), dim3(128), 0, stream>>>(
            hn16, inproj16 + (size_t)i * 1024 * 256, proj, M, 1024, 256);
        conv_silu_kernel<<<g1(M * 512), TB, 0, stream>>>(
            proj, conv_w + i * 512 * 4, conv_b + i * 512, uc, uc16, M * 512);
        gemm_wmma_kernel<3, false><<<dim3(512), dim3(32), 0, stream>>>(
            uc16, xproj16 + (size_t)i * 48 * 512, ssm, M, 48, 512);
        dt_kernel<<<g1(M * 512), TB, 0, stream>>>(
            ssm, dt_w + i * 512 * 16, dt_b + i * 512, dtb_, M * 512);
        scan_kernel<<<dim3(32), dim3(256), 0, stream>>>(
            dtb_, uc, ssm, A_log + i * 512 * 16, ysb);
        gate_kernel<<<g1(M * 512), TB, 0, stream>>>(
            ysb, uc, proj, Dp + i * 512, y16, M * 512);
        // out_proj (+residual): TDM-staged A panel, h += out_proj(y)
        gemm_wmma_lds_kernel<4, 4, true><<<dim3(512), dim3(128), 0, stream>>>(
            y16, outproj16 + (size_t)i * 256 * 512, h, M, 256, 512);
    }

    // final rmsnorm, pooling, KAN head
    rmsnormf_kernel<<<dim3(M), dim3(256), 0, stream>>>(h, normf_w, hf);
    float* out = (float*)d_out;                 // [logits(32) | emb(8192)]
    pool_kernel<<<dim3(16), dim3(256), 0, stream>>>(hf, out + 32);
    kan2_kernel<<<dim3(1), dim3(32), 0, stream>>>(out + 32, base_w2, spline_w2, spline_s2, out);
}
